// WindowAttention_84679575208277
// MI455X (gfx1250) — matrix operand order repaired and checked by reading, not compiled, not run on hardware
//
#include <hip/hip_runtime.h>

typedef __attribute__((ext_vector_type(16))) _Float16 v16h;
typedef __attribute__((ext_vector_type(8)))  _Float16 v8h;
typedef __attribute__((ext_vector_type(8)))  float    v8f;
typedef __attribute__((ext_vector_type(4)))  float    v4f;

#define B_WIN  2048
#define NTOK   49
#define CDIM   512
#define NH     16
#define HD     32
#define NPAD   64
#define ROWS   (B_WIN * NTOK)     // 100352
#define MTILES (ROWS / 16)        // 6272

__device__ __forceinline__ v8f wmma16(v16h a, v16h b, v8f c) {
  // D = A(16x32 f16) * B(32x16 f16) + C(16x16 f32)
  return __builtin_amdgcn_wmma_f32_16x16x32_f16(false, a, false, b, (short)0, c, false, false);
}

// ---------------------------------------------------------------------------
// Pre-pass A: f32 -> f16 bulk convert (8 elements / thread).
// ---------------------------------------------------------------------------
__global__ __launch_bounds__(256) void cvt_kernel(const float* __restrict__ src,
                                                  _Float16* __restrict__ dst, int n8)
{
  const int i = blockIdx.x * blockDim.x + threadIdx.x;
  if (i < n8) {
    const v4f a = *(const v4f*)(src + (size_t)i * 8);
    const v4f b = *(const v4f*)(src + (size_t)i * 8 + 4);
    v8h h;
    #pragma unroll
    for (int e = 0; e < 4; ++e) {
      h[e]     = (_Float16)a[e];
      h[e + 4] = (_Float16)b[e];
    }
    *(v8h*)(dst + (size_t)i * 8) = h;
  }
}

// ---------------------------------------------------------------------------
// Pre-pass B: materialize padded relative-position bias biasp[h][64][64]
// with mask folded in: m>=49 -> -1e30 (softmax zero), n>=49 -> 0.
// ---------------------------------------------------------------------------
__global__ __launch_bounds__(256) void bias_kernel(const float* __restrict__ rpb_table,
                                                   const int* __restrict__ rel_idx,
                                                   float* __restrict__ biasp)
{
  const int idx = blockIdx.x * blockDim.x + threadIdx.x;   // 16*64*64 = 65536
  const int h = idx >> 12;
  const int n = (idx >> 6) & 63;
  const int m = idx & 63;
  float v;
  if (m >= NTOK)      v = -1e30f;
  else if (n >= NTOK) v = 0.0f;
  else                v = rpb_table[rel_idx[n * NTOK + m] * NH + h];
  biasp[idx] = v;
}

// ---------------------------------------------------------------------------
// Kernel 1: qkv = Xh @ Wh^T + b_qkv. One wave = 16x64 tile (4 WMMA accums),
// A fragment reused across 4 N-tiles. Scatter into Qp/Kp (b,h,64,32) and
// Vt (b,h,32,64), f16, Q pre-scaled by hd^-0.5.
// ---------------------------------------------------------------------------
__global__ __launch_bounds__(256) void qkv_kernel(
    const _Float16* __restrict__ Xh, const _Float16* __restrict__ Wh,
    const float* __restrict__ b_qkv,
    _Float16* __restrict__ Qp, _Float16* __restrict__ Kp,
    _Float16* __restrict__ Vt)
{
  const int lane = threadIdx.x & 31;
  const int wv   = threadIdx.x >> 5;
  const int hl   = lane >> 4;          // lane half: 0 or 1
  const int ll   = lane & 15;
  const int mbase = blockIdx.x * 16;
  const int nb    = (blockIdx.y * 8 + wv) * 64;   // 24 tiles of 64 over 1536

  const _Float16* ar  = Xh + (size_t)(mbase + ll) * CDIM + 8 * hl;
  const _Float16* wr0 = Wh + (size_t)(nb + ll) * CDIM + 8 * hl;

  v8f acc[4] = {{}, {}, {}, {}};
  #pragma unroll
  for (int kt = 0; kt < 16; ++kt) {
    const int k0 = kt * 32;
    // A fragment: K chunks [k0+8*hl,+8) and [k0+16+8*hl,+8)
    v8h a0 = *(const v8h*)(ar + k0);
    v8h a1 = *(const v8h*)(ar + k0 + 16);
    v16h af;
    #pragma unroll
    for (int e = 0; e < 8; ++e) { af[e] = a0[e]; af[e + 8] = a1[e]; }
    #pragma unroll
    for (int j = 0; j < 4; ++j) {
      const _Float16* wr = wr0 + (size_t)j * 16 * CDIM;
      // B fragment: K chunks [k0+8*hl,+8) and [k0+16+8*hl,+8)
      v8h b0 = *(const v8h*)(wr + k0);
      v8h b1 = *(const v8h*)(wr + k0 + 16);
      v16h bf;
      #pragma unroll
      for (int e = 0; e < 8; ++e) { bf[e] = b0[e]; bf[e + 8] = b1[e]; }
      acc[j] = wmma16(af, bf, acc[j]);
    }
  }

  const float scale = 0.17677669529663687f;  // 32^-0.5
  #pragma unroll
  for (int j = 0; j < 4; ++j) {
    const int o = nb + j * 16 + ll;
    const float bias = b_qkv[o];
    const int s = o >> 9;             // 0=q 1=k 2=v
    const int h = (o >> 5) & (NH - 1);
    const int d = o & (HD - 1);
    #pragma unroll
    for (int r = 0; r < 8; ++r) {
      const int row = mbase + r + 8 * hl;    // D layout: lanes 0-15 -> M=r, 16-31 -> M=r+8
      const int bw  = row / NTOK;
      const int n   = row - bw * NTOK;
      const float v = acc[j][r] + bias;
      const size_t bh = (size_t)(bw * NH + h);
      if (s == 0)      Qp[(bh * NPAD + n) * HD + d] = (_Float16)(v * scale);
      else if (s == 1) Kp[(bh * NPAD + n) * HD + d] = (_Float16)v;
      else             Vt[(bh * HD + d) * NPAD + n] = (_Float16)v;
    }
  }
}

// ---------------------------------------------------------------------------
// Kernel 2: per (window, head): S = Q K^T + biasp, softmax, O = P V.
// One wave per (b,h); 64x64 padded S = 16 WMMA; P@V = 16 WMMA.
// ---------------------------------------------------------------------------
__global__ __launch_bounds__(128) void attn_kernel(
    const float* __restrict__ biasp,
    const _Float16* __restrict__ Qp, const _Float16* __restrict__ Kp,
    const _Float16* __restrict__ Vt, _Float16* __restrict__ Oh)
{
  __shared__ __align__(16) _Float16 Plds[4][64][72];  // 72*2=144B row stride (16B mult)
  const int lane = threadIdx.x & 31;
  const int wv   = threadIdx.x >> 5;
  const int hl   = lane >> 4;
  const int ll   = lane & 15;
  const int pair = blockIdx.x * 4 + wv;     // (b,h) index
  const int bw   = pair >> 4;
  const int h    = pair & (NH - 1);

  const _Float16* qb = Qp + (size_t)pair * NPAD * HD;
  const _Float16* kb = Kp + (size_t)pair * NPAD * HD;
  const _Float16* vb = Vt + (size_t)pair * HD * NPAD;
  const float*    bb = biasp + (size_t)h * NPAD * NPAD;

  // ---- S = Q K^T (hd=32 -> single K-step per tile) ----
  v16h qf[4], kf[4];
  #pragma unroll
  for (int i = 0; i < 4; ++i) {
    const _Float16* qr = qb + (i * 16 + ll) * HD + 8 * hl;   // A frag: two 8-chunks
    v8h q0 = *(const v8h*)qr;
    v8h q1 = *(const v8h*)(qr + 16);
    #pragma unroll
    for (int e = 0; e < 8; ++e) { qf[i][e] = q0[e]; qf[i][e + 8] = q1[e]; }
    const _Float16* kr = kb + (i * 16 + ll) * HD + 8 * hl;   // B frag: two 8-chunks
    v8h k0 = *(const v8h*)kr;
    v8h k1 = *(const v8h*)(kr + 16);
    #pragma unroll
    for (int e = 0; e < 8; ++e) { kf[i][e] = k0[e]; kf[i][e + 8] = k1[e]; }
  }
  v8f s[4][4];
  #pragma unroll
  for (int i = 0; i < 4; ++i)
    #pragma unroll
    for (int j = 0; j < 4; ++j) {
      v8f z = {};
      s[i][j] = wmma16(qf[i], kf[j], z);
    }

  // ---- add precomputed (masked) bias: coalesced loads, no branches ----
  #pragma unroll
  for (int i = 0; i < 4; ++i)
    #pragma unroll
    for (int j = 0; j < 4; ++j)
      #pragma unroll
      for (int r = 0; r < 8; ++r) {
        const int n = i * 16 + r + 8 * hl;
        const int m = j * 16 + ll;
        s[i][j][r] += bb[n * NPAD + m];
      }

  // ---- row softmax (rows live on 16-lane halves; reduce with xor 1,2,4,8) ----
  #pragma unroll
  for (int i = 0; i < 4; ++i) {
    #pragma unroll
    for (int r = 0; r < 8; ++r) {
      float mx = -1e30f;
      #pragma unroll
      for (int j = 0; j < 4; ++j) mx = fmaxf(mx, s[i][j][r]);
      #pragma unroll
      for (int off = 8; off >= 1; off >>= 1) mx = fmaxf(mx, __shfl_xor(mx, off, 32));
      float sum = 0.f;
      #pragma unroll
      for (int j = 0; j < 4; ++j) {
        float e = __expf(s[i][j][r] - mx);
        s[i][j][r] = e;
        sum += e;
      }
      #pragma unroll
      for (int off = 8; off >= 1; off >>= 1) sum += __shfl_xor(sum, off, 32);
      const float inv = 1.0f / sum;
      const int n = i * 16 + r + 8 * hl;
      #pragma unroll
      for (int j = 0; j < 4; ++j)
        Plds[wv][n][j * 16 + ll] = (_Float16)(s[i][j][r] * inv);
    }
  }

  // ---- O = P V : per i-tile, K=64 in two 32-steps, two d-tiles of 16 ----
  #pragma unroll
  for (int i = 0; i < 4; ++i) {
    v8f ov[2] = {{}, {}};
    #pragma unroll
    for (int kt = 0; kt < 2; ++kt) {
      const _Float16* pr = &Plds[wv][i * 16 + ll][kt * 32 + 8 * hl];
      v8h p0 = *(const v8h*)pr;
      v8h p1 = *(const v8h*)(pr + 16);
      v16h pf;
      #pragma unroll
      for (int e = 0; e < 8; ++e) { pf[e] = p0[e]; pf[e + 8] = p1[e]; }
      #pragma unroll
      for (int jd = 0; jd < 2; ++jd) {
        const _Float16* vr = vb + (jd * 16 + ll) * NPAD + kt * 32 + 8 * hl;
        v8h t0 = *(const v8h*)vr;
        v8h t1 = *(const v8h*)(vr + 16);
        v16h vf;
        #pragma unroll
        for (int e = 0; e < 8; ++e) { vf[e] = t0[e]; vf[e + 8] = t1[e]; }
        ov[jd] = wmma16(pf, vf, ov[jd]);
      }
    }
    #pragma unroll
    for (int jd = 0; jd < 2; ++jd)
      #pragma unroll
      for (int r = 0; r < 8; ++r) {
        const int n = i * 16 + r + 8 * hl;
        if (n < NTOK) {
          const int c = h * HD + jd * 16 + ll;
          Oh[((size_t)bw * NTOK + n) * CDIM + c] = (_Float16)ov[jd][r];
        }
      }
  }
}

// ---------------------------------------------------------------------------
// Kernel 3: out = Oh @ Wproj^T + b_proj (f32 output). One wave = 16x64 tile.
// ---------------------------------------------------------------------------
__global__ __launch_bounds__(256) void proj_kernel(
    const _Float16* __restrict__ Oh, const _Float16* __restrict__ Wh,
    const float* __restrict__ b_proj, float* __restrict__ out)
{
  const int lane = threadIdx.x & 31;
  const int wv   = threadIdx.x >> 5;
  const int hl   = lane >> 4;
  const int ll   = lane & 15;
  const int mbase = blockIdx.x * 16;
  const int nb    = wv * 64;            // 8 waves x 64 = 512 cols

  const _Float16* ar  = Oh + (size_t)(mbase + ll) * CDIM + 8 * hl;
  const _Float16* wr0 = Wh + (size_t)(nb + ll) * CDIM + 8 * hl;

  v8f acc[4] = {{}, {}, {}, {}};
  #pragma unroll
  for (int kt = 0; kt < 16; ++kt) {
    const int k0 = kt * 32;
    v8h a0 = *(const v8h*)(ar + k0);
    v8h a1 = *(const v8h*)(ar + k0 + 16);
    v16h af;
    #pragma unroll
    for (int e = 0; e < 8; ++e) { af[e] = a0[e]; af[e + 8] = a1[e]; }
    #pragma unroll
    for (int j = 0; j < 4; ++j) {
      const _Float16* wr = wr0 + (size_t)j * 16 * CDIM;
      v8h b0 = *(const v8h*)(wr + k0);
      v8h b1 = *(const v8h*)(wr + k0 + 16);
      v16h bf;
      #pragma unroll
      for (int e = 0; e < 8; ++e) { bf[e] = b0[e]; bf[e + 8] = b1[e]; }
      acc[j] = wmma16(af, bf, acc[j]);
    }
  }

  #pragma unroll
  for (int j = 0; j < 4; ++j) {
    const int o = nb + j * 16 + ll;
    const float bias = b_proj[o];
    #pragma unroll
    for (int r = 0; r < 8; ++r) {
      const int row = mbase + r + 8 * hl;
      out[(size_t)row * CDIM + o] = acc[j][r] + bias;
    }
  }
}

// ---------------------------------------------------------------------------
extern "C" void kernel_launch(void* const* d_in, const int* in_sizes, int n_in,
                              void* d_out, int out_size, void* d_ws, size_t ws_size,
                              hipStream_t stream) {
  const float* x       = (const float*)d_in[0];
  const float* rpb     = (const float*)d_in[1];
  const float* w_qkv   = (const float*)d_in[2];
  const float* b_qkv   = (const float*)d_in[3];
  const float* w_proj  = (const float*)d_in[4];
  const float* b_proj  = (const float*)d_in[5];
  const int*   rel_idx = (const int*)d_in[6];
  float* out = (float*)d_out;

  const size_t padElems = (size_t)B_WIN * NH * NPAD * HD;   // 67,108,864
  const size_t xElems   = (size_t)ROWS * CDIM;              // 51,380,224
  _Float16* Qp     = (_Float16*)d_ws;
  _Float16* Kp     = Qp + padElems;
  _Float16* Vt     = Kp + padElems;
  _Float16* Oh     = Vt + padElems;
  _Float16* Xh     = Oh + xElems;
  _Float16* Wqkvh  = Xh + xElems;
  _Float16* Wprojh = Wqkvh + (size_t)3 * CDIM * CDIM;
  float*    biasp  = (float*)(Wprojh + (size_t)CDIM * CDIM);  // 16*64*64 f32

  // Pre-passes: f16 conversions + padded bias table
  cvt_kernel<<<(int)(xElems / 8 / 256), 256, 0, stream>>>(x, Xh, (int)(xElems / 8));
  cvt_kernel<<<(3 * CDIM * CDIM / 8) / 256, 256, 0, stream>>>(w_qkv, Wqkvh, 3 * CDIM * CDIM / 8);
  cvt_kernel<<<(CDIM * CDIM / 8) / 256, 256, 0, stream>>>(w_proj, Wprojh, CDIM * CDIM / 8);
  bias_kernel<<<(NH * NPAD * NPAD) / 256, 256, 0, stream>>>(rpb, rel_idx, biasp);

  qkv_kernel<<<dim3(MTILES, 3), 256, 0, stream>>>(Xh, Wqkvh, b_qkv, Qp, Kp, Vt);
  attn_kernel<<<dim3(B_WIN * NH / 4), 128, 0, stream>>>(biasp, Qp, Kp, Vt, Oh);
  proj_kernel<<<dim3(MTILES), 256, 0, stream>>>(Oh, Wprojh, b_proj, out);
}
